// FullyParallelAttention_51754355916862
// MI455X (gfx1250) — 16-bit and 8-bit matrix operand order repaired and checked by reading, compiled, not run on hardware
//
#include <hip/hip_runtime.h>
#include <hip/hip_bf16.h>
#include <hip/hip_fp16.h>
#include <math.h>
#include <stdint.h>


typedef __attribute__((ext_vector_type(16))) int      v16i;
typedef __attribute__((ext_vector_type(8)))  float    v8f;
typedef __attribute__((ext_vector_type(16))) _Float16 v16h;
typedef __attribute__((ext_vector_type(4)))  int      v4i;

union U16i { v16i v; uint2 d2[8]; uint4 d4[4]; };
union U16h { v16h v; uint4 d4[2]; };


typedef __attribute__((address_space(1))) v4i* as1_int4p;
typedef __attribute__((address_space(3))) v4i* as3_int4p;

__device__ inline void async_copy16(void* lds, const void* glob) {
#if __has_builtin(__builtin_amdgcn_global_load_async_to_lds_b128)
  __builtin_amdgcn_global_load_async_to_lds_b128(
      (as1_int4p)(uintptr_t)glob, (as3_int4p)(uintptr_t)lds, 0, 0);
#else
  *(uint4*)lds = *(const uint4*)glob;
#endif
}

__device__ inline void async_wait() {
#if __has_builtin(__builtin_amdgcn_s_wait_asynccnt)
  __builtin_amdgcn_s_wait_asynccnt(0);
#endif
}


__device__ inline float bf16_rne(float x) {
  unsigned u = __float_as_uint(x);
  unsigned r = u + 0x7FFFu + ((u >> 16) & 1u);
  return __uint_as_float(r & 0xFFFF0000u);
}

__device__ inline unsigned char f32_to_fp8_e4m3(float x) {
  unsigned bits = __float_as_uint(x);
  unsigned char sign = (unsigned char)((bits >> 24) & 0x80u);
  float a = fabsf(x);
  if (!(a == a)) return sign;
  if (a >= 448.0f) return (unsigned char)(sign | 0x7Eu);
  if (a < 0.0009765625f) return sign;
  if (a < 0.015625f) {
    int q = (int)rintf(a * 512.0f);
    return (unsigned char)(sign | (unsigned char)q);
  }
  int e;
  float m = frexpf(a, &e);
  int E = e + 6;
  int mant = (int)rintf(m * 16.0f) - 8;
  if (mant == 8) { mant = 0; E += 1; }
  if (E > 15) return (unsigned char)(sign | 0x7Eu);
  return (unsigned char)(sign | (unsigned char)(E << 3) | (unsigned char)mant);
}

__device__ inline unsigned char quant_fp8(float v, float scale) {
  float y = v / scale;
  y = fminf(fmaxf(y, -448.0f), 448.0f);
  return f32_to_fp8_e4m3(bf16_rne(y));
}


__global__ __launch_bounds__(256) void k_f32_to_fp8(const float* __restrict__ x,
                                                    unsigned char* __restrict__ y,
                                                    size_t n) {
  for (size_t i = (size_t)blockIdx.x * blockDim.x + threadIdx.x; i < n;
       i += (size_t)gridDim.x * blockDim.x)
    y[i] = f32_to_fp8_e4m3(x[i]);
}

__global__ __launch_bounds__(256) void k_f32_to_f16(const float* __restrict__ x,
                                                    _Float16* __restrict__ y,
                                                    size_t n) {
  for (size_t i = (size_t)blockIdx.x * blockDim.x + threadIdx.x; i < n;
       i += (size_t)gridDim.x * blockDim.x)
    y[i] = (_Float16)x[i];
}


__global__ __launch_bounds__(256)
void k_ln_quant(const float* __restrict__ x, const float* __restrict__ w,
                const float* __restrict__ b, const float* __restrict__ chan_scale,
                unsigned char* __restrict__ y, int H) {
  const int row = blockIdx.x;
  const int tid = threadIdx.x;
  __shared__ float s1[8], s2[8];
  float xs[8];
  float s = 0.f, ss = 0.f;
#pragma unroll
  for (int i = 0; i < 8; ++i) {
    float v = x[(size_t)row * H + tid + i * 256];
    xs[i] = v; s += v; ss += v * v;
  }
#pragma unroll
  for (int m = 1; m < 32; m <<= 1) { s += __shfl_xor(s, m, 32); ss += __shfl_xor(ss, m, 32); }
  if ((tid & 31) == 0) { s1[tid >> 5] = s; s2[tid >> 5] = ss; }
  __syncthreads();
  float tot = 0.f, tot2 = 0.f;
#pragma unroll
  for (int i = 0; i < 8; ++i) { tot += s1[i]; tot2 += s2[i]; }
  const float mu  = tot / (float)H;
  const float var = tot2 / (float)H - mu * mu;
  const float inv = rsqrtf(var + 1e-5f);
#pragma unroll
  for (int i = 0; i < 8; ++i) {
    int c = tid + i * 256;
    float ln = (xs[i] - mu) * inv * w[c] + b[c];
    y[(size_t)row * H + c] = quant_fp8(ln, chan_scale[c]);
  }
}


__global__ __launch_bounds__(256)
void k_head_ln(_Float16* __restrict__ x, const float* __restrict__ w,
               const float* __restrict__ b, float outscale,
               int row_stride, int chunks_per_row, int col0, long nchunks) {
  long chunk = (long)blockIdx.x * 8 + (threadIdx.x >> 5);
  if (chunk >= nchunks) return;
  const int lane = threadIdx.x & 31;
  long row = chunk / chunks_per_row;
  int  cid = (int)(chunk % chunks_per_row);
  _Float16* p = x + row * row_stride + col0 + cid * 128;
  float v[4]; float s = 0.f;
#pragma unroll
  for (int i = 0; i < 4; ++i) { v[i] = (float)p[lane + i * 32]; s += v[i]; }
#pragma unroll
  for (int m = 1; m < 32; m <<= 1) s += __shfl_xor(s, m, 32);
  const float mu = s * (1.0f / 128.0f);
  float ss = 0.f;
#pragma unroll
  for (int i = 0; i < 4; ++i) { float d = v[i] - mu; ss += d * d; }
#pragma unroll
  for (int m = 1; m < 32; m <<= 1) ss += __shfl_xor(ss, m, 32);
  const float inv = rsqrtf(ss * (1.0f / 128.0f) + 1e-5f);
#pragma unroll
  for (int i = 0; i < 4; ++i) {
    int c = lane + i * 32;
    p[c] = (_Float16)(((v[i] - mu) * inv * w[c] + b[c]) * outscale);
  }
}


template <typename OutT>
__global__ __launch_bounds__(256)
void k_gemm_fp8(const unsigned char* __restrict__ A,
                const unsigned char* __restrict__ W,
                OutT* __restrict__ C, int M, int N, int K,
                const float* __restrict__ sA, const float* __restrict__ sB) {
  __shared__ unsigned char lA[2][128 * 128];
  __shared__ unsigned char lB[2][128 * 128];
  const int tid  = threadIdx.x;
  const int wave = tid >> 5, lane = tid & 31;
  const int g = lane >> 4, l16 = lane & 15;
  const int m0 = blockIdx.y * 128;
  const int n0 = blockIdx.x * 128;
  const float scale = sA[0] * sB[0];

  v8f acc[8];
#pragma unroll
  for (int i = 0; i < 8; ++i)
#pragma unroll
    for (int j = 0; j < 8; ++j) acc[i][j] = 0.f;

  auto load_tile = [&](int buf, int k0) {
#pragma unroll
    for (int i = 0; i < 4; ++i) {
      int idx = tid + i * 256;
      int row = idx >> 3;
      int col = (idx & 7) << 4;
      async_copy16(lA[buf] + row * 128 + col, A + (size_t)(m0 + row) * K + k0 + col);
      async_copy16(lB[buf] + row * 128 + col, W + (size_t)(n0 + row) * K + k0 + col);
    }
  };

  load_tile(0, 0);
  async_wait();
  __syncthreads();

  int cur = 0;
  for (int k0 = 0; k0 < K; k0 += 128) {
    if (k0 + 128 < K) {
      load_tile(cur ^ 1, k0 + 128);
      if (k0 + 256 < K)
        __builtin_prefetch(A + (size_t)(m0 + (tid >> 1)) * K + k0 + 256 + (tid & 1) * 64, 0, 1);
    }
    U16i af;
    const unsigned char* ar = lA[cur] + (wave * 16 + l16) * 128;
#pragma unroll
    for (int p = 0; p < 8; ++p) {
      int kk = (p >> 1) * 32 + g * 16 + (p & 1) * 8;
      af.d2[p] = *(const uint2*)(ar + kk);
    }
    U16i bf[2];
    {
      const unsigned char* br = lB[cur] + l16 * 128;
#pragma unroll
      for (int q = 0; q < 4; ++q) bf[0].d4[q] = *(const uint4*)(br + q * 32 + g * 16);
    }
#pragma unroll
    for (int nt = 0; nt < 8; ++nt) {
      if (nt < 7) {
        const unsigned char* br = lB[cur] + ((nt + 1) * 16 + l16) * 128;
#pragma unroll
        for (int q = 0; q < 4; ++q) bf[(nt + 1) & 1].d4[q] = *(const uint4*)(br + q * 32 + g * 16);
      }
      acc[nt] = __builtin_amdgcn_wmma_f32_16x16x128_fp8_fp8(af.v, bf[nt & 1].v, (short)0,
                                                            acc[nt], false, false);
    }
    async_wait();
    __syncthreads();
    cur ^= 1;
  }
#pragma unroll
  for (int nt = 0; nt < 8; ++nt)
#pragma unroll
    for (int r = 0; r < 8; ++r) {
      int row = m0 + wave * 16 + r + g * 8;
      int col = n0 + nt * 16 + l16;
      C[(size_t)row * N + col] = (OutT)(acc[nt][r] * scale);
    }
}


__global__ __launch_bounds__(256)
void k_gemm_f16(const _Float16* __restrict__ A, const _Float16* __restrict__ W,
                _Float16* __restrict__ C, int M, int N, int K, float scale) {
  __shared__ _Float16 lA[2][128 * 64];
  __shared__ _Float16 lB[2][128 * 64];
  const int tid  = threadIdx.x;
  const int wave = tid >> 5, lane = tid & 31;
  const int g = lane >> 4, l16 = lane & 15;
  const int m0 = blockIdx.y * 128;
  const int n0 = blockIdx.x * 128;

  v8f acc[8];
#pragma unroll
  for (int i = 0; i < 8; ++i)
#pragma unroll
    for (int j = 0; j < 8; ++j) acc[i][j] = 0.f;

  auto load_tile = [&](int buf, int k0) {
#pragma unroll
    for (int i = 0; i < 4; ++i) {
      int idx = tid + i * 256;
      int row = idx >> 3;
      int col = (idx & 7) << 3;
      async_copy16(lA[buf] + row * 64 + col, A + (size_t)(m0 + row) * K + k0 + col);
      async_copy16(lB[buf] + row * 64 + col, W + (size_t)(n0 + row) * K + k0 + col);
    }
  };

  load_tile(0, 0);
  async_wait();
  __syncthreads();

  int cur = 0;
  for (int k0 = 0; k0 < K; k0 += 64) {
    if (k0 + 64 < K) load_tile(cur ^ 1, k0 + 64);
#pragma unroll
    for (int kc = 0; kc < 2; ++kc) {
      U16h af;
      const _Float16* ar = lA[cur] + (wave * 16 + l16) * 64 + kc * 32;
      af.d4[0] = *(const uint4*)(ar + g * 8);
      af.d4[1] = *(const uint4*)(ar + 16 + g * 8);
      U16h bf[2];
      {
        const _Float16* br = lB[cur] + l16 * 64 + kc * 32 + g * 8;
        bf[0].d4[0] = *(const uint4*)(br);
        bf[0].d4[1] = *(const uint4*)(br + 16);
      }
#pragma unroll
      for (int nt = 0; nt < 8; ++nt) {
        if (nt < 7) {
          const _Float16* br = lB[cur] + ((nt + 1) * 16 + l16) * 64 + kc * 32 + g * 8;
          bf[(nt + 1) & 1].d4[0] = *(const uint4*)(br);
          bf[(nt + 1) & 1].d4[1] = *(const uint4*)(br + 16);
        }
        acc[nt] = __builtin_amdgcn_wmma_f32_16x16x32_f16(false, af.v, false, bf[nt & 1].v,
                                                         (short)0, acc[nt], false, false);
      }
    }
    async_wait();
    __syncthreads();
    cur ^= 1;
  }
#pragma unroll
  for (int nt = 0; nt < 8; ++nt)
#pragma unroll
    for (int r = 0; r < 8; ++r) {
      int row = m0 + wave * 16 + r + g * 8;
      int col = n0 + nt * 16 + l16;
      C[(size_t)row * N + col] = (_Float16)(acc[nt][r] * scale);
    }
}


__global__ __launch_bounds__(128)
void k_attn(const _Float16* __restrict__ Q, const _Float16* __restrict__ Kb,
            const _Float16* __restrict__ Vb, _Float16* __restrict__ O,
            int S, int T, int hpg,
            int q_rstride, int k_rstride, int v_rstride, int o_rstride,
            int k_col0, int v_col0, int o_col0) {
  const int h    = blockIdx.y;
  const int bidx = blockIdx.z;
  const int gsel = h / hpg;
  const _Float16* q  = Q  + (size_t)bidx * S * q_rstride + h * 128;
  const _Float16* kp = Kb + (size_t)bidx * T * k_rstride + k_col0 + gsel * 128;
  const _Float16* vp = Vb + (size_t)bidx * T * v_rstride + v_col0 + gsel * 128;
  _Float16*       op = O  + (size_t)bidx * S * o_rstride + o_col0 + h * 128;

  __shared__ _Float16 lK[32 * 128];
  __shared__ _Float16 lVt[128 * 40];
  __shared__ _Float16 lP[4][16 * 32];

  const int tid  = threadIdx.x;
  const int wave = tid >> 5, lane = tid & 31;
  const int g = lane >> 4, l16 = lane & 15;
  const int qrow0 = blockIdx.x * 64 + wave * 16;

  U16h qa[4];
  const _Float16* qr = q + (size_t)(qrow0 + l16) * q_rstride;
#pragma unroll
  for (int kc = 0; kc < 4; ++kc) {
    qa[kc].d4[0] = *(const uint4*)(qr + kc * 32 + g * 8);
    qa[kc].d4[1] = *(const uint4*)(qr + kc * 32 + 16 + g * 8);
  }

  v8f o_acc[8];
#pragma unroll
  for (int i = 0; i < 8; ++i)
#pragma unroll
    for (int j = 0; j < 8; ++j) o_acc[i][j] = 0.f;
  float mrun[8], lrun[8];
#pragma unroll
  for (int r = 0; r < 8; ++r) { mrun[r] = -1e30f; lrun[r] = 0.f; }

  for (int t0 = 0; t0 < T; t0 += 32) {
#pragma unroll
    for (int i = 0; i < 4; ++i) {
      int idx = tid + i * 128;
      int row = idx >> 4;
      int c8  = (idx & 15) * 8;
      async_copy16(lK + row * 128 + c8,
                   kp + (size_t)(t0 + row) * k_rstride + c8);
    }
#pragma unroll
    for (int i = 0; i < 4; ++i) {
      int idx = tid + i * 128;
      int row = idx >> 4;
      int c8  = (idx & 15) * 8;
      uint4 tmp = *(const uint4*)(vp + (size_t)(t0 + row) * v_rstride + c8);
      const _Float16* th = (const _Float16*)&tmp;
#pragma unroll
      for (int j = 0; j < 8; ++j) lVt[(c8 + j) * 40 + row] = th[j];
    }
    async_wait();
    __syncthreads();

    v8f s0, s1;
#pragma unroll
    for (int j = 0; j < 8; ++j) { s0[j] = 0.f; s1[j] = 0.f; }
    U16h b0[2], b1[2];
    {
      const _Float16* k0r = lK + l16 * 128 + g * 8;
      const _Float16* k1r = lK + (16 + l16) * 128 + g * 8;
      b0[0].d4[0] = *(const uint4*)(k0r); b0[0].d4[1] = *(const uint4*)(k0r + 16);
      b1[0].d4[0] = *(const uint4*)(k1r); b1[0].d4[1] = *(const uint4*)(k1r + 16);
    }
#pragma unroll
    for (int kc = 0; kc < 4; ++kc) {
      if (kc < 3) {
        const _Float16* k0r = lK + l16 * 128 + (kc + 1) * 32 + g * 8;
        const _Float16* k1r = lK + (16 + l16) * 128 + (kc + 1) * 32 + g * 8;
        b0[(kc + 1) & 1].d4[0] = *(const uint4*)(k0r);
        b0[(kc + 1) & 1].d4[1] = *(const uint4*)(k0r + 16);
        b1[(kc + 1) & 1].d4[0] = *(const uint4*)(k1r);
        b1[(kc + 1) & 1].d4[1] = *(const uint4*)(k1r + 16);
      }
      s0 = __builtin_amdgcn_wmma_f32_16x16x32_f16(false, qa[kc].v, false, b0[kc & 1].v,
                                                  (short)0, s0, false, false);
      s1 = __builtin_amdgcn_wmma_f32_16x16x32_f16(false, qa[kc].v, false, b1[kc & 1].v,
                                                  (short)0, s1, false, false);
    }

#pragma unroll
    for (int r = 0; r < 8; ++r) {
      float a = fmaxf(s0[r], s1[r]);
#pragma unroll
      for (int m = 1; m < 16; m <<= 1) a = fmaxf(a, __shfl_xor(a, m, 16));
      float mn   = fmaxf(mrun[r], a);
      float corr = __expf(mrun[r] - mn);
      float p0   = __expf(s0[r] - mn);
      float p1   = __expf(s1[r] - mn);
      float rs   = p0 + p1;
#pragma unroll
      for (int m = 1; m < 16; m <<= 1) rs += __shfl_xor(rs, m, 16);
      lrun[r] = lrun[r] * corr + rs;
      mrun[r] = mn;
#pragma unroll
      for (int nt = 0; nt < 8; ++nt) o_acc[nt][r] *= corr;
      lP[wave][(r + 8 * g) * 32 + l16]      = (_Float16)p0;
      lP[wave][(r + 8 * g) * 32 + 16 + l16] = (_Float16)p1;
    }

    U16h pf;
    const _Float16* pr = lP[wave] + l16 * 32;
    pf.d4[0] = *(const uint4*)(pr + g * 8);
    pf.d4[1] = *(const uint4*)(pr + 16 + g * 8);
    U16h vf[2];
    {
      const _Float16* vr = lVt + l16 * 40 + g * 8;
      vf[0].d4[0] = *(const uint4*)(vr);
      vf[0].d4[1] = *(const uint4*)(vr + 16);
    }
#pragma unroll
    for (int nt = 0; nt < 8; ++nt) {
      if (nt < 7) {
        const _Float16* vr = lVt + ((nt + 1) * 16 + l16) * 40 + g * 8;
        vf[(nt + 1) & 1].d4[0] = *(const uint4*)(vr);
        vf[(nt + 1) & 1].d4[1] = *(const uint4*)(vr + 16);
      }
      o_acc[nt] = __builtin_amdgcn_wmma_f32_16x16x32_f16(false, pf.v, false, vf[nt & 1].v,
                                                         (short)0, o_acc[nt], false, false);
    }
    __syncthreads();
  }

#pragma unroll
  for (int r = 0; r < 8; ++r) {
    float inv = (lrun[r] > 0.f) ? 1.f / lrun[r] : 0.f;
    int row = qrow0 + r + 8 * g;
#pragma unroll
    for (int nt = 0; nt < 8; ++nt)
      op[(size_t)row * o_rstride + nt * 16 + l16] = (_Float16)(o_acc[nt][r] * inv);
  }
}


__global__ __launch_bounds__(256)
void k_quant_smooth(const _Float16* __restrict__ x, const float* __restrict__ smooth,
                    unsigned char* __restrict__ y, size_t n, int C) {
  for (size_t i = (size_t)blockIdx.x * blockDim.x + threadIdx.x; i < n;
       i += (size_t)gridDim.x * blockDim.x) {
    int c = (int)(i % (size_t)C);
    y[i] = quant_fp8((float)x[i], smooth[c]);
  }
}


static inline unsigned char* carve(unsigned char*& p, size_t bytes) {
  unsigned char* r = p;
  p += (bytes + 255) & ~(size_t)255;
  return r;
}

extern "C" void kernel_launch(void* const* d_in, const int* in_sizes, int n_in,
                              void* d_out, int out_size, void* d_ws, size_t ws_size,
                              hipStream_t stream) {
  (void)n_in; (void)out_size; (void)ws_size;
  const int H   = in_sizes[2];
  const int BS  = in_sizes[0] / H;
  const int BSC = in_sizes[1] / H;
  const int D   = in_sizes[17];
  const int NHD = in_sizes[4] / H;
  const int GD  = in_sizes[10] / H;
  const int NH  = NHD / D, G = GD / D, HPG = NH / G;
  const int B = 2, S = BS / B, SC = BSC / B;
  const int CAT = 2 * NHD;

  const float* hidden   = (const float*)d_in[0];
  const float* cond     = (const float*)d_in[1];
  const float* ln_w     = (const float*)d_in[2];
  const float* ln_b     = (const float*)d_in[3];
  const float* wq       = (const float*)d_in[4];
  const float* wq_s     = (const float*)d_in[5];
  const float* q_in_s   = (const float*)d_in[6];
  const float* wqx      = (const float*)d_in[7];
  const float* wqx_s    = (const float*)d_in[8];
  const float* wk       = (const float*)d_in[10];
  const float* wk_s     = (const float*)d_in[11];
  const float* k_in_s   = (const float*)d_in[12];
  const float* wv       = (const float*)d_in[13];
  const float* wv_s     = (const float*)d_in[14];
  const float* v_in_s   = (const float*)d_in[15];
  const float* w_kv_x   = (const float*)d_in[16];
  const float* qln_w    = (const float*)d_in[17];
  const float* qln_b    = (const float*)d_in[18];
  const float* kln_w    = (const float*)d_in[19];
  const float* kln_b    = (const float*)d_in[20];
  const float* qxln_w   = (const float*)d_in[21];
  const float* qxln_b   = (const float*)d_in[22];
  const float* kxln_w   = (const float*)d_in[23];
  const float* kxln_b   = (const float*)d_in[24];
  const float* wproj    = (const float*)d_in[25];
  const float* wproj_s  = (const float*)d_in[26];
  const float* proj_in  = (const float*)d_in[27];
  const float* smooth   = (const float*)d_in[28];
  float* out = (float*)d_out;

  unsigned char* ws = (unsigned char*)d_ws;
  unsigned char* ln8    = carve(ws, (size_t)BS * H);
  unsigned char* wq8    = carve(ws, (size_t)NHD * H);
  unsigned char* wqx8   = carve(ws, (size_t)NHD * H);
  unsigned char* wk8    = carve(ws, (size_t)GD * H);
  unsigned char* wv8    = carve(ws, (size_t)GD * H);
  unsigned char* wp8    = carve(ws, (size_t)H * CAT);
  _Float16* qf   = (_Float16*)carve(ws, (size_t)BS * NHD * 2);
  _Float16* qxf  = (_Float16*)carve(ws, (size_t)BS * NHD * 2);
  _Float16* kf   = (_Float16*)carve(ws, (size_t)BS * GD * 2);
  _Float16* vf   = (_Float16*)carve(ws, (size_t)BS * GD * 2);
  _Float16* c16  = (_Float16*)carve(ws, (size_t)BSC * H * 2);
  _Float16* wkv16= (_Float16*)carve(ws, (size_t)(2 * GD) * H * 2);
  _Float16* kvx  = (_Float16*)carve(ws, (size_t)BSC * (2 * GD) * 2);
  _Float16* cat  = (_Float16*)carve(ws, (size_t)BS * CAT * 2);
  unsigned char* cat8 = carve(ws, (size_t)BS * CAT);

  k_f32_to_fp8<<<1024, 256, 0, stream>>>(wq,    wq8,  (size_t)NHD * H);
  k_f32_to_fp8<<<1024, 256, 0, stream>>>(wqx,   wqx8, (size_t)NHD * H);
  k_f32_to_fp8<<<1024, 256, 0, stream>>>(wk,    wk8,  (size_t)GD * H);
  k_f32_to_fp8<<<1024, 256, 0, stream>>>(wv,    wv8,  (size_t)GD * H);
  k_f32_to_fp8<<<1024, 256, 0, stream>>>(wproj, wp8,  (size_t)H * CAT);
  k_f32_to_f16<<<1024, 256, 0, stream>>>(cond,   c16,   (size_t)BSC * H);
  k_f32_to_f16<<<1024, 256, 0, stream>>>(w_kv_x, wkv16, (size_t)(2 * GD) * H);

  k_ln_quant<<<BS, 256, 0, stream>>>(hidden, ln_w, ln_b, q_in_s, ln8, H);

  k_gemm_fp8<_Float16><<<dim3(NHD / 128, BS / 128), 256, 0, stream>>>(
      ln8, wq8, qf, BS, NHD, H, q_in_s, wq_s);
  k_gemm_fp8<_Float16><<<dim3(NHD / 128, BS / 128), 256, 0, stream>>>(
      ln8, wqx8, qxf, BS, NHD, H, q_in_s, wqx_s);
  k_gemm_fp8<_Float16><<<dim3(GD / 128, BS / 128), 256, 0, stream>>>(
      ln8, wk8, kf, BS, GD, H, k_in_s, wk_s);
  k_gemm_fp8<_Float16><<<dim3(GD / 128, BS / 128), 256, 0, stream>>>(
      ln8, wv8, vf, BS, GD, H, v_in_s, wv_s);

  k_gemm_f16<<<dim3((2 * GD) / 128, BSC / 128), 256, 0, stream>>>(
      c16, wkv16, kvx, BSC, 2 * GD, H, 1.0f);

  const float qscale = 1.0f / sqrtf((float)D);
  k_head_ln<<<(BS * NH) / 8, 256, 0, stream>>>(qf,  qln_w,  qln_b,  qscale, NHD, NH, 0, (long)BS * NH);
  k_head_ln<<<(BS * NH) / 8, 256, 0, stream>>>(qxf, qxln_w, qxln_b, qscale, NHD, NH, 0, (long)BS * NH);
  k_head_ln<<<(BS * G) / 8, 256, 0, stream>>>(kf,  kln_w,  kln_b,  1.0f, GD, G, 0, (long)BS * G);
  k_head_ln<<<(BSC * G) / 8, 256, 0, stream>>>(kvx, kxln_w, kxln_b, 1.0f, 2 * GD, G, 0, (long)BSC * G);

  k_attn<<<dim3(S / 64, NH, B), 128, 0, stream>>>(
      qf, kf, vf, cat, S, S, HPG, NHD, GD, GD, CAT, 0, 0, 0);
  k_attn<<<dim3(S / 64, NH, B), 128, 0, stream>>>(
      qxf, kvx, kvx, cat, S, SC, HPG, NHD, 2 * GD, 2 * GD, CAT, 0, GD, NHD);

  k_quant_smooth<<<2048, 256, 0, stream>>>(cat, smooth, cat8, (size_t)BS * CAT, CAT);
  k_gemm_fp8<float><<<dim3(H / 128, BS / 128), 256, 0, stream>>>(
      cat8, wp8, out, BS, H, CAT, proj_in, wproj_s);
}
